// ToMe_FSTW_PEMF_50946902065315
// MI455X (gfx1250) — matrix operand order repaired and checked by reading, not compiled, not run on hardware
//
#include <hip/hip_runtime.h>
#include <hip/hip_bf16.h>

// ---------------------------------------------------------------------------
// Problem constants (from the reference)
// ---------------------------------------------------------------------------
#define BATCH  64
#define P0     729
#define CDIM   1152
#define HDIM   3584
#define NHEADS 16
#define CH     72          // CDIM / NHEADS
#define TGT    128
#define MTOT   (BATCH*TGT) // 8192 rows into the MLP

typedef __attribute__((ext_vector_type(16))) _Float16 v16h;
typedef __attribute__((ext_vector_type(8)))  float    v8f;
typedef __attribute__((ext_vector_type(4)))  int      v4i;

// ---------------------------------------------------------------------------
// 1) x + spatial_pos, and size = 1
// ---------------------------------------------------------------------------
__global__ __launch_bounds__(256) void addpos_kernel(
    const float* __restrict__ x, const float* __restrict__ pos,
    float* __restrict__ out, float* __restrict__ size0)
{
  size_t idx = (size_t)blockIdx.x * 256 + threadIdx.x;
  size_t pc = idx % ((size_t)P0 * CDIM);
  out[idx] = x[idx] + pos[pc];
  if ((idx % CDIM) == 0) size0[idx / CDIM] = 1.0f;
}

// ---------------------------------------------------------------------------
// 2) metric = normalize(mean over heads).  One wave32 per (b, token).
// ---------------------------------------------------------------------------
__global__ void metric_kernel(const float* __restrict__ x,
                              float* __restrict__ metric, int t)
{
  int blk = blockIdx.x;
  int b = blk / t, tok = blk % t;
  int lane = threadIdx.x;
  const float* row = x + ((size_t)b * t + tok) * CDIM;
  float mv[3];
  int n = 0;
  float ss = 0.f;
  for (int c = lane; c < CH; c += 32) {
    float s = 0.f;
    #pragma unroll
    for (int h = 0; h < NHEADS; ++h) s += row[h * CH + c];
    s *= (1.0f / NHEADS);
    mv[n++] = s;
    ss += s * s;
  }
  for (int m = 16; m >= 1; m >>= 1) ss += __shfl_xor(ss, m, 32);
  float inv = 1.0f / sqrtf(ss);
  float* mo = metric + ((size_t)b * t + tok) * CH;
  n = 0;
  for (int c = lane; c < CH; c += 32) mo[c] = mv[n++] * inv;
}

// ---------------------------------------------------------------------------
// 3) node_max / node_idx: for each even-row i, max_j dot(a_i, b_j)
//    (argmax with first-index tie break, matching jnp.argmax)
// ---------------------------------------------------------------------------
__global__ __launch_bounds__(256) void scores_kernel(
    const float* __restrict__ metric, float* __restrict__ node_max,
    int* __restrict__ node_idx, int t, int na, int nb)
{
  __shared__ float av[CH];
  __shared__ float rv[256];
  __shared__ int   ri[256];
  int b = blockIdx.x / na;
  int i = blockIdx.x % na;
  int tid = threadIdx.x;
  const float* mb = metric + (size_t)b * t * CH;
  if (tid < CH) av[tid] = mb[(size_t)(2 * i) * CH + tid];
  __syncthreads();
  float best = -3.0e38f;
  int bj = 0x7fffffff;
  for (int j = tid; j < nb; j += 256) {
    const float* br = mb + (size_t)(2 * j + 1) * CH;
    float s = 0.f;
    for (int c = 0; c < CH; ++c) s += av[c] * br[c];
    if (s > best || (s == best && j < bj)) { best = s; bj = j; }
  }
  rv[tid] = best; ri[tid] = bj;
  __syncthreads();
  for (int off = 128; off > 0; off >>= 1) {
    if (tid < off) {
      float o = rv[tid + off]; int oi = ri[tid + off];
      if (o > rv[tid] || (o == rv[tid] && oi < ri[tid])) { rv[tid] = o; ri[tid] = oi; }
    }
    __syncthreads();
  }
  if (tid == 0) { node_max[b * 365 + i] = rv[0]; node_idx[b * 365 + i] = ri[0]; }
}

// ---------------------------------------------------------------------------
// 4) stable descending argsort of node_max (bitonic, tie -> lower index),
//    then split into unm/src and gather dst = node_idx[src].
// ---------------------------------------------------------------------------
__global__ __launch_bounds__(512) void sort_select_kernel(
    const float* __restrict__ node_max, const int* __restrict__ node_idx,
    int* __restrict__ unm_idx, int* __restrict__ src_idx, int* __restrict__ dst_idx,
    int na, int r)
{
  __shared__ float key[512];
  __shared__ int   idx[512];
  int b = blockIdx.x;
  int tid = threadIdx.x;
  key[tid] = (tid < na) ? node_max[b * 365 + tid] : -3.0e38f;
  idx[tid] = tid;
  __syncthreads();
  for (int k = 2; k <= 512; k <<= 1) {
    for (int j = k >> 1; j > 0; j >>= 1) {
      int ixj = tid ^ j;
      if (ixj > tid) {
        float ka = key[tid], kb = key[ixj];
        int ia = idx[tid], ib = idx[ixj];
        bool aBeforeB = (ka > kb) || (ka == kb && ia < ib);
        bool doSwap = ((tid & k) == 0) ? !aBeforeB : aBeforeB;
        if (doSwap) { key[tid] = kb; key[ixj] = ka; idx[tid] = ib; idx[ixj] = ia; }
      }
      __syncthreads();
    }
  }
  if (tid < r) {
    int s = idx[tid];
    src_idx[b * 365 + tid] = s;
    dst_idx[b * 365 + tid] = node_idx[b * 365 + s];
  } else if (tid < na) {
    unm_idx[b * 365 + (tid - r)] = idx[tid];
  }
}

// ---------------------------------------------------------------------------
// 5) merge: out rows = [src[unm], dst + scattered src]; deterministic
// ---------------------------------------------------------------------------
__global__ __launch_bounds__(256) void merge_kernel(
    const float* __restrict__ xin, const float* __restrict__ szin,
    float* __restrict__ xout, float* __restrict__ szout,
    const int* __restrict__ unm_idx, const int* __restrict__ src_idx,
    const int* __restrict__ dst_idx, int t, int r, int nb, int u)
{
  __shared__ int sdst[384];
  __shared__ int ssrc[384];
  int o = blockIdx.x;
  int b = blockIdx.y;
  int tid = threadIdx.x;
  int outT = u + nb;
  const float* xb = xin + (size_t)b * t * CDIM;
  float* xo = xout + ((size_t)b * outT + o) * CDIM;

  if (o < u) {  // unmerged: value unchanged (x*s/s)
    int s2 = 2 * unm_idx[b * 365 + o];
    const float* src = xb + (size_t)s2 * CDIM;
    for (int c = tid; c < CDIM; c += 256) xo[c] = src[c];
    if (tid == 0) szout[b * outT + o] = szin[b * t + s2];
    return;
  }
  int j = o - u;
  for (int kk = tid; kk < r; kk += 256) {
    sdst[kk] = dst_idx[b * 365 + kk];
    ssrc[kk] = src_idx[b * 365 + kk];
  }
  __syncthreads();
  float szb = szin[b * t + 2 * j + 1];
  float news = szb;
  float acc[5];
  {
    const float* base = xb + (size_t)(2 * j + 1) * CDIM;
    int q = 0;
    for (int c = tid; c < CDIM; c += 256) acc[q++] = base[c] * szb;
  }
  for (int kk = 0; kk < r; ++kk) {
    if (sdst[kk] == j) {           // uniform across block
      int s2 = 2 * ssrc[kk];
      float szk = szin[b * t + s2];
      news += szk;
      const float* srow = xb + (size_t)s2 * CDIM;
      int q = 0;
      for (int c = tid; c < CDIM; c += 256) acc[q++] += srow[c] * szk;
    }
  }
  float inv = 1.0f / news;
  int q = 0;
  for (int c = tid; c < CDIM; c += 256) xo[c] = acc[q++] * inv;
  if (tid == 0) szout[b * outT + o] = news;
}

// ---------------------------------------------------------------------------
// 6) casts: f32 -> f16, and transposed f32 (K,N) -> f16 (N,K)
// ---------------------------------------------------------------------------
__global__ __launch_bounds__(256) void cast_f16_kernel(
    const float* __restrict__ in, _Float16* __restrict__ out, size_t n)
{
  size_t i = (size_t)blockIdx.x * 256 + threadIdx.x;
  if (i < n) out[i] = (_Float16)in[i];
}

__global__ __launch_bounds__(256) void transcast_kernel(
    const float* __restrict__ w, _Float16* __restrict__ wt, int K, int N)
{
  __shared__ float tile[32][33];
  int tx = threadIdx.x & 31, ty = threadIdx.x >> 5;
  int n0 = blockIdx.x * 32, k0 = blockIdx.y * 32;
  #pragma unroll
  for (int i = 0; i < 32; i += 8)
    tile[ty + i][tx] = w[(size_t)(k0 + ty + i) * N + (n0 + tx)];
  __syncthreads();
  #pragma unroll
  for (int i = 0; i < 32; i += 8)
    wt[(size_t)(n0 + ty + i) * K + (k0 + tx)] = (_Float16)tile[tx][ty + i];
}

// ---------------------------------------------------------------------------
// 7) WMMA GEMM: C(MxN) = A(MxK,f16) * BT(NxK,f16)^T (+bias [,GELU])
//    Block tile 128x128, 8 waves arranged 4x2, wave tile 32x64.
//    BK = 64; LDS rows padded to 72 halfs -> conflict-free ds_load_b128.
//    Global -> LDS staging uses gfx1250 async-to-LDS when available.
// ---------------------------------------------------------------------------
#define LDSP 72   // padded row stride in halfs (144 B): banks r*36 mod 64

#if defined(__HIP_DEVICE_COMPILE__) && __has_builtin(__builtin_amdgcn_global_load_async_to_lds_b128)
#define USE_ASYNC_LDS 1
#elif defined(__HIP_DEVICE_COMPILE__) && __has_builtin(__builtin_amdgcn_global_load_lds)
#define USE_LDS_DMA 1
#endif

__device__ inline void stage16(_Float16* lds, const _Float16* g) {
#if defined(USE_ASYNC_LDS)
  // param 1: global (addrspace(1), prints as __device__) pointer to int4
  // param 2: LDS (addrspace(3)) pointer to int4
  __builtin_amdgcn_global_load_async_to_lds_b128(
      (__attribute__((address_space(1))) v4i*)(g),
      (__attribute__((address_space(3))) v4i*)(lds), 0, 0);
#elif defined(USE_LDS_DMA)
  __builtin_amdgcn_global_load_lds(
      (const __attribute__((address_space(1))) v4i*)(g),
      (__attribute__((address_space(3))) v4i*)(lds), 16, 0, 0);
#else
  *(float4*)lds = *(const float4*)g;
#endif
}

__device__ inline void stage_wait() {
#if defined(USE_ASYNC_LDS)
  asm volatile("s_wait_asynccnt 0" ::: "memory");
#elif defined(USE_LDS_DMA)
  asm volatile("s_wait_loadcnt 0" ::: "memory");
#endif
}

__device__ inline v8f wmma_f16(v16h a, v16h b, v8f c) {
  return __builtin_amdgcn_wmma_f32_16x16x32_f16(
      /*neg_a=*/false, a, /*neg_b=*/false, b,
      /*c_mod=*/(short)0, c, /*reuse_a=*/false, /*reuse_b=*/false);
}

// A fragment (16x32 f16): lane L (m=L%16, half=L/16) holds
// K = half*8 + [0..7] and 16 + half*8 + [0..7] -> two 16B chunks of a row.
__device__ inline v16h ld_fragA(const _Float16* rowp, int h8) {
  union { v16h v; float4 q[2]; } u;
  u.q[0] = *(const float4*)(rowp + h8);
  u.q[1] = *(const float4*)(rowp + 16 + h8);
  return u.v;
}
// B fragment (32x16 f16) with BT stored [n][k]: lane L (n=L%16, half=L/16)
// holds K = half*16 + [0..15] -> one contiguous 32B run.
__device__ inline v16h ld_fragB(const _Float16* p) {
  union { v16h v; float4 q[2]; } u;
  u.q[0] = *(const float4*)(p);
  u.q[1] = *(const float4*)(p + 8);
  return u.v;
}

template <bool GELU, typename OutT>
__global__ __launch_bounds__(256) void gemm_wmma_kernel(
    const _Float16* __restrict__ A, const _Float16* __restrict__ BT,
    const float* __restrict__ bias, OutT* __restrict__ out,
    int M, int N, int K)
{
  __shared__ _Float16 As[128 * LDSP];
  __shared__ _Float16 Bs[128 * LDSP];
  const int tid  = threadIdx.x;
  const int lane = tid & 31;
  const int wave = tid >> 5;
  const int gm0 = blockIdx.y * 128;
  const int gn0 = blockIdx.x * 128;
  const int wm = (wave & 3) * 32;   // 0,32,64,96
  const int wn = (wave >> 2) * 64;  // 0,64

  // staging: each thread owns half a row (32 halfs = 64B = 4x16B) of A and B
  const int srow = tid >> 1;           // 0..127
  const int scol = (tid & 1) * 32;     // 0,32
  const _Float16* Ag = A  + (size_t)(gm0 + srow) * K + scol;
  const _Float16* Bg = BT + (size_t)(gn0 + srow) * K + scol;
  _Float16* Al = As + srow * LDSP + scol;
  _Float16* Bl = Bs + srow * LDSP + scol;

  v8f acc[2][4] = {};

  const int mrow = lane & 15;
  const int h8  = (lane >> 4) * 8;
  const int k16 = (lane >> 4) * 16;

  for (int k0 = 0; k0 < K; k0 += 64) {
    __syncthreads();
    #pragma unroll
    for (int c = 0; c < 4; ++c) {
      stage16(Al + c * 8, Ag + k0 + c * 8);
      stage16(Bl + c * 8, Bg + k0 + c * 8);
    }
#if !defined(USE_ASYNC_LDS) && !defined(USE_LDS_DMA)
    if (k0 + 64 < K) {  // gfx1250 global_prefetch_b8 on next K tile
      __builtin_prefetch(Ag + k0 + 64, 0, 3);
      __builtin_prefetch(Bg + k0 + 64, 0, 3);
    }
#endif
    stage_wait();
    __syncthreads();
    #pragma unroll
    for (int kh = 0; kh < 2; ++kh) {
      const int ko = kh * 32;
      v16h b0 = ld_fragA(Bs + (wn      + mrow) * LDSP + ko, h8);
      v16h b1 = ld_fragA(Bs + (wn + 16 + mrow) * LDSP + ko, h8);
      v16h b2 = ld_fragA(Bs + (wn + 32 + mrow) * LDSP + ko, h8);
      v16h b3 = ld_fragA(Bs + (wn + 48 + mrow) * LDSP + ko, h8);
      #pragma unroll
      for (int mi = 0; mi < 2; ++mi) {
        v16h a = ld_fragA(As + (wm + mi * 16 + mrow) * LDSP + ko, h8);
        acc[mi][0] = wmma_f16(a, b0, acc[mi][0]);
        acc[mi][1] = wmma_f16(a, b1, acc[mi][1]);
        acc[mi][2] = wmma_f16(a, b2, acc[mi][2]);
        acc[mi][3] = wmma_f16(a, b3, acc[mi][3]);
      }
    }
  }

  const int col = lane & 15;
  const int ro  = (lane >> 4) * 8;  // C/D layout: lanes 16-31 hold M = i+8
  auto store_tile = [&](v8f c, int m0, int n0) {
    int n  = gn0 + n0 + col;
    int mb = gm0 + m0 + ro;
    float bv = bias[n];
    #pragma unroll
    for (int i = 0; i < 8; ++i) {
      float v = c[i] + bv;
      if (GELU) v = 0.5f * v * (1.0f + erff(v * 0.70710678118654752f));
      out[(size_t)(mb + i) * N + n] = (OutT)v;
    }
  };
  #pragma unroll
  for (int mi = 0; mi < 2; ++mi)
    #pragma unroll
    for (int ni = 0; ni < 4; ++ni)
      store_tile(acc[mi][ni], wm + mi * 16, wn + ni * 16);
}

// ---------------------------------------------------------------------------
// Launcher
// ---------------------------------------------------------------------------
extern "C" void kernel_launch(void* const* d_in, const int* in_sizes, int n_in,
                              void* d_out, int out_size, void* d_ws, size_t ws_size,
                              hipStream_t stream) {
  const float* x    = (const float*)d_in[0];
  const float* spos = (const float*)d_in[1];
  const float* w1   = (const float*)d_in[2];
  const float* b1   = (const float*)d_in[3];
  const float* w2   = (const float*)d_in[4];
  const float* b2   = (const float*)d_in[5];
  float* outp = (float*)d_out;

  // bump allocator on workspace
  char* wp = (char*)d_ws;
  auto alloc = [&](size_t bytes) -> void* {
    void* r = (void*)wp;
    wp += (bytes + 255) & ~(size_t)255;
    return r;
  };
  float* xb0 = (float*)alloc((size_t)BATCH * P0 * CDIM * 4);   // 729-token buf
  float* xb1 = (float*)alloc((size_t)BATCH * 365 * CDIM * 4);  // 365-token buf
  float* s0  = (float*)alloc((size_t)BATCH * P0 * 4);
  float* s1  = (float*)alloc((size_t)BATCH * 365 * 4);
  float* metric   = (float*)alloc((size_t)BATCH * P0 * CH * 4);
  float* node_max = (float*)alloc((size_t)BATCH * 365 * 4);
  int*   node_idx = (int*)alloc((size_t)BATCH * 365 * 4);
  int*   unm_idx  = (int*)alloc((size_t)BATCH * 365 * 4);
  int*   src_idx  = (int*)alloc((size_t)BATCH * 365 * 4);
  int*   dst_idx  = (int*)alloc((size_t)BATCH * 365 * 4);
  _Float16* A16  = (_Float16*)alloc((size_t)MTOT * CDIM * 2);
  _Float16* W1T  = (_Float16*)alloc((size_t)HDIM * CDIM * 2);
  _Float16* W2T  = (_Float16*)alloc((size_t)HDIM * HDIM * 2);
  _Float16* H16  = (_Float16*)alloc((size_t)MTOT * HDIM * 2);

  // 1) positional embed + size init
  {
    size_t nt = (size_t)BATCH * P0 * CDIM;
    addpos_kernel<<<(int)(nt / 256), 256, 0, stream>>>(x, spos, xb0, s0);
  }

  // 2) three ToMe merge rounds: 729 -> 365 -> 183 -> 128
  const int ts[3] = {729, 365, 183};
  const int rs[3] = {364, 182, 55};
  float* curX[4]  = {xb0, xb1, xb0, xb1};
  float* curS[4]  = {s0, s1, s0, s1};
  for (int it = 0; it < 3; ++it) {
    int t = ts[it], r = rs[it];
    int na = (t + 1) / 2, nb = t / 2;
    int u = na - r;
    float* cx = curX[it];  float* cs = curS[it];
    float* ox = curX[it + 1]; float* os = curS[it + 1];
    metric_kernel<<<BATCH * t, 32, 0, stream>>>(cx, metric, t);
    scores_kernel<<<BATCH * na, 256, 0, stream>>>(metric, node_max, node_idx, t, na, nb);
    sort_select_kernel<<<BATCH, 512, 0, stream>>>(node_max, node_idx,
                                                  unm_idx, src_idx, dst_idx, na, r);
    merge_kernel<<<dim3(u + nb, BATCH), 256, 0, stream>>>(cx, cs, ox, os,
                                                          unm_idx, src_idx, dst_idx,
                                                          t, r, nb, u);
  }
  float* xfinal = curX[3];  // (64,128,1152) in xb1

  // 3) casts for WMMA
  {
    size_t n = (size_t)MTOT * CDIM;
    cast_f16_kernel<<<(int)((n + 255) / 256), 256, 0, stream>>>(xfinal, A16, n);
  }
  transcast_kernel<<<dim3(HDIM / 32, CDIM / 32), 256, 0, stream>>>(w1, W1T, CDIM, HDIM);
  transcast_kernel<<<dim3(HDIM / 32, HDIM / 32), 256, 0, stream>>>(w2, W2T, HDIM, HDIM);

  // 4) GEMM1 + GELU -> H16 ; GEMM2 + bias -> out (f32)
  gemm_wmma_kernel<true, _Float16>
      <<<dim3(HDIM / 128, MTOT / 128), 256, 0, stream>>>(A16, W1T, b1, H16,
                                                         MTOT, HDIM, CDIM);
  gemm_wmma_kernel<false, float>
      <<<dim3(HDIM / 128, MTOT / 128), 256, 0, stream>>>(H16, W2T, b2, outp,
                                                         MTOT, HDIM, HDIM);
}
